// SelfAttention1D_64415919505984
// MI455X (gfx1250) — matrix operand order repaired and checked by reading, not compiled, not run on hardware
//
#include <hip/hip_runtime.h>
#include <hip/hip_bf16.h>

// CDNA5 / gfx1250 flash-attention style SelfAttention1D.
// bf16 WMMA (v_wmma_f32_16x16x32_bf16) everywhere, f32 accumulation.

typedef __attribute__((ext_vector_type(16))) __bf16 v16bf;
typedef __attribute__((ext_vector_type(8)))  float  v8f;

#define B_ 8
#define C_ 256
#define T_ 2048
#define R_ 32
#define XTS 264  // LDS x^T tile row stride in halves (528B, 16B-aligned, bank-skewed)

// ---- helpers -------------------------------------------------------------

__device__ __forceinline__ unsigned short f2bf(float f) {
  unsigned u = __float_as_uint(f);
  unsigned r = 0x7FFFu + ((u >> 16) & 1u);
  if ((u & 0x7F800000u) == 0x7F800000u) r = 0;  // inf/nan: truncate
  return (unsigned short)((u + r) >> 16);
}

// B-operand load: lane holds 16 contiguous K-values of one N column
// (contiguous k order used throughout this file, not the ISA table order; wave32).
__device__ __forceinline__ v16bf loadB16(const unsigned short* p) {
  union { v16bf v; uint4 u[2]; } r;
  r.u[0] = *(const uint4*)(p);
  r.u[1] = *(const uint4*)(p + 8);
  return r.v;
}

// A-operand load: lane holds K = 2*off..2*off+15 (contiguous) of one M row
// (lanes 0-15 off=0, lanes 16-31 off=8; matches loadB16 K order).
__device__ __forceinline__ v16bf loadA16(const unsigned short* p, int off) {
  union { v16bf v; uint4 u[2]; } r;
  r.u[0] = *(const uint4*)(p + 2 * off);
  r.u[1] = *(const uint4*)(p + 2 * off + 8);
  return r.v;
}

__device__ __forceinline__ v8f wmma_bf16(v16bf a, v16bf b, v8f c) {
  // (neg_a, A, neg_b, B, c_mod, C, reuse_a, reuse_b)
  return __builtin_amdgcn_wmma_f32_16x16x32_bf16(false, a, false, b,
                                                 (short)0, c, false, false);
}

// ---- kernel 0: f32 weights -> bf16 ---------------------------------------

__global__ void cvt_kernel(const float* __restrict__ src,
                           unsigned short* __restrict__ dst, int n) {
  int i = blockIdx.x * blockDim.x + threadIdx.x;
  if (i < n) dst[i] = f2bf(src[i]);
}

// ---- kernel 1: projections q,k [B,T,32] bf16 ; v^T [B,256,T] bf16 --------

__global__ __launch_bounds__(256) void proj_kernel(
    const float* __restrict__ x,
    const unsigned short* __restrict__ Wq, const float* __restrict__ bq,
    const unsigned short* __restrict__ Wk, const float* __restrict__ bk,
    const unsigned short* __restrict__ Wv, const float* __restrict__ bv,
    unsigned short* __restrict__ qO, unsigned short* __restrict__ kO,
    unsigned short* __restrict__ vTO) {
  __shared__ unsigned short lds_x[64 * XTS];  // x^T tile [64 t][256 c] bf16

  const int tid = threadIdx.x;
  const int b   = blockIdx.x >> 5;          // 32 t-slices per batch
  const int t0  = (blockIdx.x & 31) * 64;

  // stage + transpose x[b][:, t0:t0+64] into LDS as bf16 (coalesced over t)
  #pragma unroll 1
  for (int e = 0; e < 64; ++e) {
    int idx = e * 256 + tid;
    int c = idx >> 6;
    int tt = idx & 63;
    lds_x[tt * XTS + c] = f2bf(x[((size_t)b * C_ + c) * T_ + t0 + tt]);
  }
  __syncthreads();

  const int w = tid >> 5, lane = tid & 31;
  const int n    = lane & 15;               // N column / M row within tile
  const int offA = (lane < 16) ? 0 : 8;     // A-layout K-group / C-layout M-group
  const int offB = (lane < 16) ? 0 : 16;    // B-layout K-group
  const int ttile = w & 3;
  const unsigned short* arow = &lds_x[(ttile * 16 + n) * XTS];

  // ---- V projection: D[t,o] = x^T W_v^T ; each wave: 1 t-tile x 8 o-tiles
  {
    const int o_base = (w >> 2) * 128;
    v8f acc[8];
    #pragma unroll
    for (int j = 0; j < 8; ++j)
      #pragma unroll
      for (int i = 0; i < 8; ++i) acc[j][i] = 0.0f;

    #pragma unroll 1
    for (int cc = 0; cc < 8; ++cc) {        // 8 chunks of K=32 over C
      v16bf a = loadA16(arow, cc * 32 + offA);
      #pragma unroll
      for (int jj = 0; jj < 2; ++jj) {      // multi-buffer: 4 B tiles in flight
        v16bf bm[4];
        #pragma unroll
        for (int u = 0; u < 4; ++u)
          bm[u] = loadB16(
              &Wv[(size_t)(o_base + (jj * 4 + u) * 16 + n) * C_ + cc * 32 + offB]);
        #pragma unroll
        for (int u = 0; u < 4; ++u)
          acc[jj * 4 + u] = wmma_bf16(a, bm[u], acc[jj * 4 + u]);
      }
    }
    // add bias, store transposed v^T[b][o][t] as bf16 (8 contiguous t per lane)
    #pragma unroll
    for (int j = 0; j < 8; ++j) {
      int o = o_base + j * 16 + n;
      float bias = bv[o];
      union { unsigned short s[8]; uint4 u; } pk;
      #pragma unroll
      for (int i = 0; i < 8; ++i) pk.s[i] = f2bf(acc[j][i] + bias);
      *(uint4*)&vTO[((size_t)b * C_ + o) * T_ + t0 + ttile * 16 + offA] = pk.u;
    }
  }

  // ---- Q / K projections: waves 0-3 -> Q, waves 4-7 -> K
  {
    const bool isQ = (w < 4);
    const unsigned short* W = isQ ? Wq : Wk;
    const float* bias = isQ ? bq : bk;
    unsigned short* dst = isQ ? qO : kO;

    v8f acc[2];
    #pragma unroll
    for (int rt = 0; rt < 2; ++rt)
      #pragma unroll
      for (int i = 0; i < 8; ++i) acc[rt][i] = 0.0f;

    #pragma unroll 1
    for (int cc = 0; cc < 8; ++cc) {
      v16bf a = loadA16(arow, cc * 32 + offA);
      v16bf bm[2];
      #pragma unroll
      for (int rt = 0; rt < 2; ++rt)
        bm[rt] = loadB16(&W[(size_t)(rt * 16 + n) * C_ + cc * 32 + offB]);
      #pragma unroll
      for (int rt = 0; rt < 2; ++rt)
        acc[rt] = wmma_bf16(a, bm[rt], acc[rt]);
    }
    #pragma unroll
    for (int rt = 0; rt < 2; ++rt) {
      int r = rt * 16 + n;
      float bs = bias[r];
      #pragma unroll
      for (int i = 0; i < 8; ++i)
        dst[((size_t)b * T_ + t0 + ttile * 16 + offA + i) * R_ + r] =
            f2bf(acc[rt][i] + bs);
    }
  }
}

// ---- kernel 2: flash attention, one 16-query tile per wave ---------------

__global__ __launch_bounds__(128) void attn_kernel(
    const float* __restrict__ x,
    const unsigned short* __restrict__ qB,   // [B,T,32] bf16
    const unsigned short* __restrict__ kB,   // [B,T,32] bf16
    const unsigned short* __restrict__ vT,   // [B,256,T] bf16
    const float* __restrict__ gamma,
    float* __restrict__ out) {
  __shared__ float          lds_S[4][16 * 128];   // logits, wave-private
  __shared__ unsigned short lds_P[4][16 * 128];   // probs bf16, wave-private
  __shared__ float          lds_red[4][32];       // pairwise max/sum exchange
  __shared__ float          lds_corr[4][16];
  __shared__ float          lds_linv[4][16];

  const int tid = threadIdx.x;
  const int w = tid >> 5, lane = tid & 31;
  const int n    = lane & 15;
  const int offA = (lane < 16) ? 0 : 8;
  const int offB = (lane < 16) ? 0 : 16;

  const int qt = blockIdx.x * 4 + w;    // global query tile
  const int b = qt >> 7;                // 128 tiles per batch (same b for 4 waves)
  const int t_base = (qt & 127) * 16;

  // Q tile, A-layout resident for the whole pass
  v16bf qa = loadA16(&qB[((size_t)b * T_ + t_base + n) * R_], offA);

  v8f acc[16];                          // O accumulator: 16 rows x 256 ch
  #pragma unroll
  for (int ot = 0; ot < 16; ++ot)
    #pragma unroll
    for (int i = 0; i < 8; ++i) acc[ot][i] = 0.0f;

  float m_run = -3.0e38f, l_run = 0.0f; // row stats (identical in a lane pair)

  float*          Swave = lds_S[w];
  unsigned short* Pwave = lds_P[w];

  #pragma unroll 1
  for (int kc = 0; kc < 16; ++kc) {     // 128 keys per chunk
    const int s_base = kc * 128;

    // prefetch next chunk's K (8KB) and V (64KB) region into caches:
    // 128 threads cover 128 K rows and 256 V rows (2 x 128B lines each).
    if (kc + 1 < 16) {
      const int ns = (kc + 1) * 128;
      __builtin_prefetch(&kB[((size_t)b * T_ + ns + tid) * R_], 0, 0);
      const unsigned short* vp0 = &vT[((size_t)b * C_ + (tid << 1)) * T_ + ns];
      const unsigned short* vp1 = &vT[((size_t)b * C_ + (tid << 1) + 1) * T_ + ns];
      __builtin_prefetch(vp0, 0, 0);
      __builtin_prefetch(vp0 + 64, 0, 0);
      __builtin_prefetch(vp1, 0, 0);
      __builtin_prefetch(vp1 + 64, 0, 0);
    }

    // S[16,128] = Q K^T  (K=32 in one WMMA per 16-key subtile), 4 B tiles in flight
    #pragma unroll
    for (int jj = 0; jj < 2; ++jj) {
      v16bf kb[4];
      #pragma unroll
      for (int u = 0; u < 4; ++u)
        kb[u] = loadB16(
            &kB[((size_t)b * T_ + s_base + (jj * 4 + u) * 16 + n) * R_ + offB]);
      #pragma unroll
      for (int u = 0; u < 4; ++u) {
        v8f z;
        #pragma unroll
        for (int i = 0; i < 8; ++i) z[i] = 0.0f;
        v8f sv = wmma_bf16(qa, kb[u], z);
        #pragma unroll
        for (int i = 0; i < 8; ++i)
          Swave[(offA + i) * 128 + (jj * 4 + u) * 16 + n] = sv[i];
      }
    }

    // online softmax: lane pair (2r,2r+1) owns query row r; 64 cols per lane.
    // Same-wave LDS ops are in order -> no barrier needed for the exchanges.
    {
      const int row = lane >> 1, half = lane & 1;
      float* Srow = &Swave[row * 128 + half * 64];
      unsigned short* Prow = &Pwave[row * 128 + half * 64];
      float cm = -3.0e38f;
      for (int c2 = 0; c2 < 64; ++c2) cm = fmaxf(cm, Srow[c2]);
      lds_red[w][lane] = cm;
      float mo = lds_red[w][lane ^ 1];
      float mn = fmaxf(fmaxf(cm, mo), m_run);
      float corr = __expf(m_run - mn);
      float psum = 0.0f;
      for (int c2 = 0; c2 < 64; ++c2) {
        float p = __expf(Srow[c2] - mn);
        psum += p;
        Prow[c2] = f2bf(p);
      }
      lds_red[w][lane] = psum;
      psum += lds_red[w][lane ^ 1];
      l_run = l_run * corr + psum;      // identical in both lanes of the pair
      m_run = mn;
      if (half == 0) lds_corr[w][row] = corr;
    }

    // rescale O by per-row correction (rows offA..offA+7 in C/D layout)
    float c8[8];
    #pragma unroll
    for (int i = 0; i < 8; ++i) c8[i] = lds_corr[w][offA + i];
    #pragma unroll
    for (int ot = 0; ot < 16; ++ot)
      #pragma unroll
      for (int i = 0; i < 8; ++i) acc[ot][i] *= c8[i];

    // O += P V : 4 K=32 subchunks x 16 o-tiles, 4 B tiles in flight
    #pragma unroll
    for (int ss = 0; ss < 4; ++ss) {
      v16bf pa = loadA16(&Pwave[n * 128 + ss * 32], offA);
      #pragma unroll
      for (int og = 0; og < 4; ++og) {
        v16bf vb[4];
        #pragma unroll
        for (int u = 0; u < 4; ++u)
          vb[u] = loadB16(&vT[((size_t)b * C_ + (og * 4 + u) * 16 + n) * T_ +
                              s_base + ss * 32 + offB]);
        #pragma unroll
        for (int u = 0; u < 4; ++u)
          acc[og * 4 + u] = wmma_bf16(pa, vb[u], acc[og * 4 + u]);
      }
    }
  }

  // finalize: out = x + gamma * (O / l) ; C/D layout -> 8 contiguous t/lane
  if ((lane & 1) == 0) lds_linv[w][lane >> 1] = 1.0f / l_run;
  float inv8[8];
  #pragma unroll
  for (int i = 0; i < 8; ++i) inv8[i] = lds_linv[w][offA + i];
  const float g = gamma[0];
  #pragma unroll
  for (int ot = 0; ot < 16; ++ot) {
    size_t base = ((size_t)b * C_ + ot * 16 + n) * T_ + t_base + offA;
    #pragma unroll
    for (int i = 0; i < 8; ++i)
      out[base + i] = x[base + i] + g * acc[ot][i] * inv8[i];
  }
}

// ---- host launcher --------------------------------------------------------

extern "C" void kernel_launch(void* const* d_in, const int* in_sizes, int n_in,
                              void* d_out, int out_size, void* d_ws, size_t ws_size,
                              hipStream_t stream) {
  const float* x     = (const float*)d_in[0];
  const float* Wq    = (const float*)d_in[1];
  const float* bq    = (const float*)d_in[2];
  const float* Wk    = (const float*)d_in[3];
  const float* bk    = (const float*)d_in[4];
  const float* Wv    = (const float*)d_in[5];
  const float* bv    = (const float*)d_in[6];
  const float* gamma = (const float*)d_in[7];
  float* out = (float*)d_out;

  // workspace layout (bf16 staging, ~10.2 MB total, 256B-aligned blocks)
  unsigned char* w = (unsigned char*)d_ws;
  unsigned short* wq_bf = (unsigned short*)(w + 0);              //  16 KB
  unsigned short* wk_bf = (unsigned short*)(w + 16384);          //  16 KB
  unsigned short* wv_bf = (unsigned short*)(w + 32768);          // 128 KB
  unsigned short* q_bf  = (unsigned short*)(w + 163840);         //   1 MB
  unsigned short* k_bf  = (unsigned short*)(w + 163840 + 1048576);
  unsigned short* vT_bf = (unsigned short*)(w + 163840 + 2097152); // 8 MB

  cvt_kernel<<<(R_ * C_ + 255) / 256, 256, 0, stream>>>(Wq, wq_bf, R_ * C_);
  cvt_kernel<<<(R_ * C_ + 255) / 256, 256, 0, stream>>>(Wk, wk_bf, R_ * C_);
  cvt_kernel<<<(C_ * C_ + 255) / 256, 256, 0, stream>>>(Wv, wv_bf, C_ * C_);

  proj_kernel<<<B_ * (T_ / 64), 256, 0, stream>>>(
      x, wq_bf, bq, wk_bf, bk, wv_bf, bv, q_bf, k_bf, vT_bf);

  attn_kernel<<<(B_ * T_ / 16) / 4, 128, 0, stream>>>(
      x, q_bf, k_bf, vT_bf, gamma, out);

  (void)in_sizes; (void)n_in; (void)out_size; (void)ws_size;
}
